// AMS_86861418594720
// MI455X (gfx1250) — compile-verified, matrix operand order checked by reading, not run on hardware
//
#include <hip/hip_runtime.h>
#include <math.h>

// ---------------- fixed model config ----------------
constexpr int Bn = 32, WINn = 96, Nn = 64, Dn = 64, DFFn = 256;
constexpr int Hn = 8, EHn = 32, KHn = 8, NEn = 4;
constexpr int TOK = Bn * WINn * Nn;                  // 196608 tokens
constexpr size_t XEL = (size_t)TOK * Dn;             // 12582912 elements
constexpr int BND = Bn * Nn * Dn;                    // 131072
constexpr int CHUNK = 24576;                         // FFN token chunk (8 chunks)

typedef __attribute__((ext_vector_type(16))) __bf16 v16bf;
typedef __attribute__((ext_vector_type(8)))  __bf16 v8bf;
typedef __attribute__((ext_vector_type(8)))  float  v8f;

#define DEV static __device__ __forceinline__

DEV float geluf(float x) { return 0.5f * x * (1.0f + erff(x * 0.70710678118654752f)); }

// ---------- fragment loads (no bounds checks; K%32==0 guaranteed) ----------
// CDNA5 16-bit A layout (16x32 tile): lanes 0-15 row M=lane, K = k0+[0..8) and
// k0+16+[0..8); lanes 16-31: +8 on both K bases.
DEV v16bf frag_a_fastf(const float* __restrict__ prow, int ka)
{
    const float4* pa = (const float4*)(prow + ka);
    const float4* pb = (const float4*)(prow + ka + 16);
    float4 a0 = pa[0], a1 = pa[1], b0 = pb[0], b1 = pb[1];
    v16bf f;
    f[0]=(__bf16)a0.x; f[1]=(__bf16)a0.y; f[2]=(__bf16)a0.z; f[3]=(__bf16)a0.w;
    f[4]=(__bf16)a1.x; f[5]=(__bf16)a1.y; f[6]=(__bf16)a1.z; f[7]=(__bf16)a1.w;
    f[8]=(__bf16)b0.x; f[9]=(__bf16)b0.y; f[10]=(__bf16)b0.z; f[11]=(__bf16)b0.w;
    f[12]=(__bf16)b1.x; f[13]=(__bf16)b1.y; f[14]=(__bf16)b1.z; f[15]=(__bf16)b1.w;
    return f;
}

DEV v16bf frag_b_fastt(const __bf16* __restrict__ prow, int ka)
{
    v8bf u = *(const v8bf*)(prow + ka);
    v8bf v = *(const v8bf*)(prow + ka + 16);
    v16bf f;
#pragma unroll
    for (int j = 0; j < 8; ++j) { f[j] = u[j]; f[j + 8] = v[j]; }
    return f;
}

DEV v16bf frag_b_fastf(const float* __restrict__ Bf, int col, int N, int ka)
{
    v16bf f;
#pragma unroll
    for (int j = 0; j < 8; ++j) {
        f[j]     = (__bf16)Bf[(size_t)(ka + j) * N + col];
        f[j + 8] = (__bf16)Bf[(size_t)(ka + 16 + j) * N + col];
    }
    return f;
}

// ---------------- fast GEMM: wave computes 32x32 C; 4 wmma per K-step ----------------
// Requires M%32==0, N%32==0, K%32==0 (true for every call site; intra_lin K is
// zero-padded to a multiple of 32). Block = 256 threads = 8 waves arranged
// 4(M) x 2(N): block tile 128x64.
template <int BL>
__global__ void gemm_wmma_fast(const float* __restrict__ A, const void* __restrict__ Bop,
                               const float* __restrict__ bias, const float* __restrict__ res,
                               float* __restrict__ C,
                               int M, int N, int K,
                               long long sA, long long sB, long long sC, long long sR,
                               int act)
{
    const int lane = threadIdx.x & 31;
    const int wave = threadIdx.x >> 5;
    const int m0 = blockIdx.x * 128 + (wave >> 1) * 32;
    const int n0 = blockIdx.y * 64 + (wave & 1) * 32;
    if (m0 >= M || n0 >= N) return;           // wave-uniform: EXEC all-ones at WMMA
    const int batch = blockIdx.z;
    A += (size_t)batch * sA;
    C += (size_t)batch * sC;
    const __bf16* Bt = (const __bf16*)Bop + ((BL == 0) ? (size_t)batch * sB : 0);
    const float*  Bf = (const float*)Bop  + ((BL == 1) ? (size_t)batch * sB : 0);
    if (res) res += (size_t)batch * sR;

    const int r0 = m0 + (lane & 15);
    const int c0 = n0 + (lane & 15);
    const int half = (lane >> 4) << 3;
    const float* pA0 = A + (size_t)r0 * K;
    const float* pA1 = A + (size_t)(r0 + 16) * K;
    const __bf16* pB0 = Bt + (size_t)c0 * K;
    const __bf16* pB1 = Bt + (size_t)(c0 + 16) * K;

    v8f acc00 = {}, acc01 = {}, acc10 = {}, acc11 = {};
    for (int k0 = 0; k0 < K; k0 += 32) {
        const int ka = k0 + half;
        __builtin_prefetch(pA0 + k0 + 32, 0, 1);   // speculative -> global_prefetch_b8
        __builtin_prefetch(pA1 + k0 + 32, 0, 1);
        v16bf a0 = frag_a_fastf(pA0, ka);
        v16bf a1 = frag_a_fastf(pA1, ka);
        v16bf b0, b1;
        if (BL == 0) { b0 = frag_b_fastt(pB0, ka); b1 = frag_b_fastt(pB1, ka); }
        else         { b0 = frag_b_fastf(Bf, c0, N, ka); b1 = frag_b_fastf(Bf, c0 + 16, N, ka); }
        acc00 = __builtin_amdgcn_wmma_f32_16x16x32_bf16(false, a0, false, b0, (short)0, acc00, false, false);
        acc01 = __builtin_amdgcn_wmma_f32_16x16x32_bf16(false, a0, false, b1, (short)0, acc01, false, false);
        acc10 = __builtin_amdgcn_wmma_f32_16x16x32_bf16(false, a1, false, b0, (short)0, acc10, false, false);
        acc11 = __builtin_amdgcn_wmma_f32_16x16x32_bf16(false, a1, false, b1, (short)0, acc11, false, false);
    }
    // C/D layout per tile: VGPR v -> row tile_m0 + v + (lane>=16 ? 8 : 0), col tile_n0 + (lane&15)
    const int rb = m0 + ((lane >> 4) << 3);
#pragma unroll
    for (int t = 0; t < 4; ++t) {
        const v8f* accp = (t == 0) ? &acc00 : (t == 1) ? &acc01 : (t == 2) ? &acc10 : &acc11;
        const int row0 = rb + ((t >> 1) ? 16 : 0);
        const int col  = n0 + (lane & 15) + ((t & 1) ? 16 : 0);
        const float bv = bias ? bias[col] : 0.0f;
#pragma unroll
        for (int v = 0; v < 8; ++v) {
            int row = row0 + v;
            float val = (*accp)[v] + bv;
            if (act) val = geluf(val);
            if (res) val += res[(size_t)row * N + col];
            C[(size_t)row * N + col] = val;
        }
    }
}

// Weight prep: W[K,N] fp32 -> Wt[N,Kp] bf16 (transpose + cast + zero-pad K to Kp)
__global__ void wprep(const float* __restrict__ W, __bf16* __restrict__ Wt,
                      int K, int N, int Kp)
{
    int i = blockIdx.x * blockDim.x + threadIdx.x;
    if (i >= N * Kp) return;
    int n = i / Kp, k = i % Kp;
    Wt[i] = (k < K) ? (__bf16)W[(size_t)k * N + n] : (__bf16)0.0f;
}

// Top-2 gating + importance accumulation
__global__ void gating_kernel(const float* __restrict__ x, const float* __restrict__ wg,
                              float* __restrict__ gates, float* __restrict__ imp)
{
    int t = blockIdx.x * blockDim.x + threadIdx.x;
    if (t >= TOK) return;
    const float* xr = x + (size_t)t * Dn;
    float lg[NEn];
#pragma unroll
    for (int e = 0; e < NEn; ++e) {
        float s = 0.0f;
        for (int d = 0; d < Dn; ++d) s += xr[d] * wg[d * NEn + e];
        lg[e] = s;
    }
    float m1 = fmaxf(fmaxf(lg[0], lg[1]), fmaxf(lg[2], lg[3]));
    float m2 = -1e30f; bool used = false;
#pragma unroll
    for (int e = 0; e < NEn; ++e) {
        if (!used && lg[e] == m1) { used = true; continue; }
        m2 = fmaxf(m2, lg[e]);
    }
    float den = 0.0f, g[NEn];
#pragma unroll
    for (int e = 0; e < NEn; ++e) {
        float v = (lg[e] >= m2) ? __expf(lg[e] - m1) : 0.0f;
        g[e] = v; den += v;
    }
    int wn = t % (WINn * Nn);
#pragma unroll
    for (int e = 0; e < NEn; ++e) {
        float v = g[e] / den;
        gates[(size_t)t * NEn + e] = v;
        atomicAdd(&imp[wn * NEn + e], v);
    }
}

__global__ void balance_kernel(const float* __restrict__ imp, float* __restrict__ loss)
{
    __shared__ float red[256];
    const int NELT = WINn * Nn * NEn;
    int tid = threadIdx.x;
    float s = 0.0f;
    for (int i = tid; i < NELT; i += 256) s += imp[i];
    red[tid] = s; __syncthreads();
    for (int o = 128; o > 0; o >>= 1) { if (tid < o) red[tid] += red[tid + o]; __syncthreads(); }
    float mean = red[0] / (float)NELT;
    __syncthreads();
    float sq = 0.0f;
    for (int i = tid; i < NELT; i += 256) { float d = imp[i] - mean; sq += d * d; }
    red[tid] = sq; __syncthreads();
    for (int o = 128; o > 0; o >>= 1) { if (tid < o) red[tid] += red[tid + o]; __syncthreads(); }
    if (tid == 0) {
        float var = red[0] / (float)(NELT - 1);
        atomicAdd(loss, 0.01f * var / (mean * mean + 1e-10f));
    }
}

// Hypergraph incidence: A = softmax(top-8 of node_emb @ edge_emb^T); also A^T and c_loss
__global__ void hyper_A_kernel(const float* __restrict__ node_emb, const float* __restrict__ edge_emb,
                               float* __restrict__ Aout, float* __restrict__ At,
                               float* __restrict__ loss)
{
    int n = threadIdx.x;
    if (n >= Nn) return;
    float lg[EHn];
    for (int e = 0; e < EHn; ++e) {
        float s = 0.0f;
        for (int d = 0; d < Dn; ++d) s += node_emb[n * Dn + d] * edge_emb[e * Dn + d];
        lg[e] = s;
    }
    float tmp[EHn];
    for (int e = 0; e < EHn; ++e) tmp[e] = lg[e];
    float kth = -1e30f;
    for (int i = 0; i < KHn; ++i) {
        int bi = 0; float bv = -1e30f;
        for (int e = 0; e < EHn; ++e) if (tmp[e] > bv) { bv = tmp[e]; bi = e; }
        kth = bv; tmp[bi] = -1e30f;
    }
    float mx = -1e30f;
    for (int e = 0; e < EHn; ++e) if (lg[e] >= kth) mx = fmaxf(mx, lg[e]);
    float den = 0.0f, a[EHn];
    for (int e = 0; e < EHn; ++e) {
        float v = (lg[e] >= kth) ? __expf(lg[e] - mx) : 0.0f;
        a[e] = v; den += v;
    }
    float ssq = 0.0f;
    for (int e = 0; e < EHn; ++e) {
        float v = a[e] / den;
        Aout[n * EHn + e] = v;
        At[e * Nn + n] = v;
        ssq += v * v;
    }
    atomicAdd(loss, 0.01f * ssq / (float)(Nn * EHn));   // c_loss * LOSS_COEF1
}

// xi[(b,n,p,s),d] = xs[b, p*ps+s, n, d]
__global__ void permute_xi(const float* __restrict__ xs, float* __restrict__ xi, int ps, int pn)
{
    size_t i = (size_t)blockIdx.x * blockDim.x + threadIdx.x;
    if (i >= XEL) return;
    int d = (int)(i & 63);
    size_t r = i >> 6;
    int s = (int)(r % ps); r /= ps;
    int p = (int)(r % pn); r /= pn;
    int n = (int)(r % Nn); int b = (int)(r / Nn);
    int w = p * ps + s;
    xi[i] = xs[(((size_t)(b * WINn + w) * Nn) + n) * Dn + d];
}

// Padding transpose: io2[(b,n),d,p<Kp] = (p < pn) ? io[(b,n),p,d] : 0
__global__ void tr_io_pad(const float* __restrict__ io, float* __restrict__ io2,
                          int pn, int Kp)
{
    size_t tot = (size_t)Bn * Nn * Dn * Kp;
    size_t i = (size_t)blockIdx.x * blockDim.x + threadIdx.x;
    if (i >= tot) return;
    int p = (int)(i % Kp); size_t r = i / Kp;
    int d = (int)(r % Dn); size_t bn = r / Dn;
    io2[i] = (p < pn) ? io[(bn * pn + p) * Dn + d] : 0.0f;
}

// Intra-patch attention: learned q (Lq=1) over ps<=8 keys, H=8 heads, dk=8
__global__ void intra_attn(const float* __restrict__ q, const float* __restrict__ kk,
                           const float* __restrict__ vv, float* __restrict__ io,
                           int ps, int pn)
{
    int bn = blockIdx.x;
    int t = threadIdx.x;
    if (t >= pn * Hn) return;
    int p = t / Hn, h = t % Hn;
    int n = bn % Nn;
    const float* qr = q + ((size_t)(n * pn + p)) * Dn + h * 8;
    size_t base = ((size_t)bn * pn + p) * ps;
    float sc[8]; float mx = -1e30f;
    for (int s = 0; s < ps; ++s) {
        const float* kr = kk + (base + s) * Dn + h * 8;
        float d = 0.0f;
#pragma unroll
        for (int e = 0; e < 8; ++e) d += qr[e] * kr[e];
        d *= 0.35355339059327373f;            // 1/sqrt(8)
        sc[s] = d; mx = fmaxf(mx, d);
    }
    float den = 0.0f;
    for (int s = 0; s < ps; ++s) { sc[s] = __expf(sc[s] - mx); den += sc[s]; }
    float o[8] = {0,0,0,0,0,0,0,0};
    for (int s = 0; s < ps; ++s) {
        const float* vr = vv + (base + s) * Dn + h * 8;
        float a = sc[s] / den;
#pragma unroll
        for (int e = 0; e < 8; ++e) o[e] += a * vr[e];
    }
    float* orow = io + ((size_t)bn * pn + p) * Dn + h * 8;
#pragma unroll
    for (int e = 0; e < 8; ++e) orow[e] = o[e];
}

// Inter-patch attention over pn<=48 tokens, per (b,n,head)
__global__ void inter_attn(const float* __restrict__ q2, const float* __restrict__ k2,
                           const float* __restrict__ v2, float* __restrict__ eo,
                           int pn, int dm)
{
    int bn = blockIdx.x;
    int h = blockIdx.y;
    int l = threadIdx.x;
    if (l >= pn) return;
    int dki = dm / Hn;
    size_t rb = (size_t)bn * pn;
    const float* qr = q2 + (rb + l) * dm + h * dki;
    float scl = rsqrtf((float)dki);
    float sc[48]; float mx = -1e30f;
    for (int s = 0; s < pn; ++s) {
        const float* kr = k2 + (rb + s) * dm + h * dki;
        float d = 0.0f;
        for (int e = 0; e < dki; ++e) d += qr[e] * kr[e];
        d *= scl;
        sc[s] = d; mx = fmaxf(mx, d);
    }
    float den = 0.0f;
    for (int s = 0; s < pn; ++s) { sc[s] = __expf(sc[s] - mx); den += sc[s]; }
    float* orow = eo + (rb + l) * dm + h * dki;
    for (int e = 0; e < dki; ++e) {
        float o = 0.0f;
        for (int s = 0; s < pn; ++s) o += sc[s] * v2[(rb + s) * dm + h * dki + e];
        orow[e] = o / den;
    }
}

// out = LN1(xs + intra + inter) ; intraC is [B,N,D,WIN], interF is [(b,n,p), s*D+d]
__global__ void combine_ln1(const float* __restrict__ xs, const float* __restrict__ intraC,
                            const float* __restrict__ interF,
                            const float* __restrict__ g1, const float* __restrict__ b1,
                            float* __restrict__ x1, int ps, int pn)
{
    int t = blockIdx.x * blockDim.x + threadIdx.x;
    if (t >= TOK) return;
    int n = t % Nn; int bw = t / Nn;
    int w = bw % WINn; int b = bw / WINn;
    int p = w / ps, s = w % ps;
    size_t xoff = (size_t)t * Dn;
    size_t ibase = ((size_t)(b * Nn + n) * Dn) * WINn + w;
    size_t eoff = (((size_t)(b * Nn + n) * pn + p) * ps + s) * Dn;
    float buf[Dn]; float mean = 0.0f;
    for (int d = 0; d < Dn; ++d) {
        float v = xs[xoff + d] + intraC[ibase + (size_t)d * WINn] + interF[eoff + d];
        buf[d] = v; mean += v;
    }
    mean *= (1.0f / Dn);
    float var = 0.0f;
    for (int d = 0; d < Dn; ++d) { float dd = buf[d] - mean; var += dd * dd; }
    var *= (1.0f / Dn);
    float inv = rsqrtf(var + 1e-5f);
    for (int d = 0; d < Dn; ++d)
        x1[xoff + d] = (buf[d] - mean) * inv * g1[d] + b1[d];
}

// d_out += gate_e * LN2(x1 + ffn)
__global__ void final_ln2_acc(const float* __restrict__ x1, const float* __restrict__ ffn,
                              const float* __restrict__ g2, const float* __restrict__ b2,
                              const float* __restrict__ gates, float* __restrict__ out, int e)
{
    int t = blockIdx.x * blockDim.x + threadIdx.x;
    if (t >= TOK) return;
    size_t off = (size_t)t * Dn;
    float buf[Dn]; float mean = 0.0f;
    for (int d = 0; d < Dn; ++d) { float v = x1[off + d] + ffn[off + d]; buf[d] = v; mean += v; }
    mean *= (1.0f / Dn);
    float var = 0.0f;
    for (int d = 0; d < Dn; ++d) { float dd = buf[d] - mean; var += dd * dd; }
    var *= (1.0f / Dn);
    float inv = rsqrtf(var + 1e-5f);
    float g = gates[(size_t)t * NEn + e];
    for (int d = 0; d < Dn; ++d)
        out[off + d] += g * ((buf[d] - mean) * inv * g2[d] + b2[d]);
}

extern "C" void kernel_launch(void* const* d_in, const int* in_sizes, int n_in,
                              void* d_out, int out_size, void* d_ws, size_t ws_size,
                              hipStream_t stream)
{
    (void)in_sizes; (void)n_in; (void)out_size; (void)ws_size;
    const float* x  = (const float*)d_in[0];
    const float* wg = (const float*)d_in[1];
    float* out = (float*)d_out;
    float* lossp = out + XEL;
    static const int PATCH[4] = {8, 6, 4, 2};

    // ---- workspace carve-up ----
    char* wp = (char*)d_ws;
    auto alloc = [&](size_t elems) -> float* {
        float* p = (float*)wp;
        wp += ((elems * sizeof(float)) + 255) & ~(size_t)255;
        return p;
    };
    float* gates = alloc((size_t)TOK * NEn);
    float* imp   = alloc((size_t)WINn * Nn * NEn);
    float* Ainc  = alloc(Nn * EHn);
    float* At    = alloc(Nn * EHn);
    float* qproj = alloc((size_t)Nn * 48 * Dn);
    float* edge  = alloc((size_t)Bn * WINn * EHn * Dn);
    float* bufA  = alloc(XEL);
    float* bufB  = alloc(XEL);
    float* bufC  = alloc(XEL);
    float* bufD  = alloc(XEL);
    float* bufE  = alloc(XEL);
    float* bufG  = alloc(XEL);
    float* bufI1 = alloc((size_t)Bn * Nn * Dn * 64);   // holds intra O-proj output [B*N*pn, D]
    float* bufI2 = alloc((size_t)Bn * Nn * 48 * Dn);
    float* bufH  = alloc((size_t)CHUNK * DFFn);
    auto allocbf = [&](size_t elems) -> __bf16* {
        __bf16* p = (__bf16*)wp;
        wp += ((elems * sizeof(__bf16)) + 255) & ~(size_t)255;
        return p;
    };
    __bf16* Wt_sp = allocbf(64 * 64);
    __bf16* Wt_iq = allocbf(64 * 64);
    __bf16* Wt_ik = allocbf(64 * 64);
    __bf16* Wt_iv = allocbf(64 * 64);
    __bf16* Wt_io = allocbf(64 * 64);
    __bf16* Wt_q2 = allocbf(512 * 512);
    __bf16* Wt_k2 = allocbf(512 * 512);
    __bf16* Wt_v2 = allocbf(512 * 512);
    __bf16* Wt_o2 = allocbf(512 * 512);
    __bf16* Wt_ln = allocbf(96 * 64);                  // padded [WIN, Kp<=64]
    __bf16* Wt_f1 = allocbf(64 * 256);
    __bf16* Wt_f2 = allocbf(256 * 64);

    // fast GEMM: aligned shapes (M%32==0, N%32==0, K%32==0), bf16 Wt weights
    auto GEMMF = [&](const float* A, const __bf16* Bt, const float* bias,
                     const float* res, float* C, int M, int N, int K, int act) {
        dim3 grid((M + 127) / 128, (N + 63) / 64, 1);
        gemm_wmma_fast<0><<<grid, 256, 0, stream>>>(A, Bt, bias, res, C, M, N, K,
                                                    0, 0, 0, 0, act);
    };
    // fast batched GEMM with fp32 strided B (hypergraph per-slice matmuls)
    auto GEMMB = [&](const float* A, const float* Bf, float* C, int M, int N, int K,
                     long long sA, long long sB, long long sC, int batch) {
        dim3 grid((M + 127) / 128, (N + 63) / 64, batch);
        gemm_wmma_fast<1><<<grid, 256, 0, stream>>>(A, Bf, nullptr, nullptr, C, M, N, K,
                                                    sA, sB, sC, 0, 0);
    };
    auto WPREP = [&](const float* W, __bf16* Wt, int K, int N, int Kp) {
        int tot = N * Kp;
        wprep<<<(tot + 255) / 256, 256, 0, stream>>>(W, Wt, K, N, Kp);
    };

    hipMemsetAsync(imp, 0, (size_t)WINn * Nn * NEn * sizeof(float), stream);
    hipMemsetAsync(lossp, 0, sizeof(float), stream);
    hipMemcpyAsync(out, x, XEL * sizeof(float), hipMemcpyDeviceToDevice, stream);

    gating_kernel<<<(TOK + 255) / 256, 256, 0, stream>>>(x, wg, gates, imp);
    balance_kernel<<<1, 256, 0, stream>>>(imp, lossp);

    for (int e = 0; e < 4; ++e) {
        const int ps = PATCH[e], pn = WINn / ps, dm = Dn * ps;
        const int Kp = (pn + 31) & ~31;       // intra_lin K padded to 32/64
        void* const* P = d_in + 2 + e * 31;
        const float* node_emb = (const float*)P[0];
        const float* edge_emb = (const float*)P[1];
        const float* spW = (const float*)P[2];  const float* spB = (const float*)P[3];
        const float* quer = (const float*)P[4];
        const float* iqW = (const float*)P[5];  const float* iqB = (const float*)P[6];
        const float* ikW = (const float*)P[7];  const float* ikB = (const float*)P[8];
        const float* ivW = (const float*)P[9];  const float* ivB = (const float*)P[10];
        const float* ioW = (const float*)P[11]; const float* ioB = (const float*)P[12];
        const float* q2W = (const float*)P[13]; const float* q2B = (const float*)P[14];
        const float* k2W = (const float*)P[15]; const float* k2B = (const float*)P[16];
        const float* v2W = (const float*)P[17]; const float* v2B = (const float*)P[18];
        const float* o2W = (const float*)P[19]; const float* o2B = (const float*)P[20];
        const float* lnW = (const float*)P[21]; const float* lnB = (const float*)P[22];
        const float* f1W = (const float*)P[23]; const float* f1B = (const float*)P[24];
        const float* f2W = (const float*)P[25]; const float* f2B = (const float*)P[26];
        const float* n1g = (const float*)P[27]; const float* n1b = (const float*)P[28];
        const float* n2g = (const float*)P[29]; const float* n2b = (const float*)P[30];

        // weight prep (transpose+cast to bf16; intra_lin zero-padded to Kp)
        WPREP(spW, Wt_sp, Dn, Dn, Dn);
        WPREP(iqW, Wt_iq, Dn, Dn, Dn); WPREP(ikW, Wt_ik, Dn, Dn, Dn);
        WPREP(ivW, Wt_iv, Dn, Dn, Dn); WPREP(ioW, Wt_io, Dn, Dn, Dn);
        WPREP(q2W, Wt_q2, dm, dm, dm); WPREP(k2W, Wt_k2, dm, dm, dm);
        WPREP(v2W, Wt_v2, dm, dm, dm); WPREP(o2W, Wt_o2, dm, dm, dm);
        WPREP(lnW, Wt_ln, pn, WINn, Kp);
        WPREP(f1W, Wt_f1, Dn, DFFn, Dn);
        WPREP(f2W, Wt_f2, DFFn, Dn, DFFn);

        // hypergraph incidence + c_loss
        hyper_A_kernel<<<1, 64, 0, stream>>>(node_emb, edge_emb, Ainc, At, lossp);
        // edge[b,w] = A^T(EHxN) * x_slice(NxD), batched over B*WIN
        GEMMB(At, x, edge, EHn, Dn, Nn, 0, (long long)Nn * Dn, (long long)EHn * Dn, Bn * WINn);
        // t2[b,w] = A(NxEH) * edge_slice(EHxD) -> bufA
        GEMMB(Ainc, edge, bufA, Nn, Dn, EHn, 0, (long long)EHn * Dn, (long long)Nn * Dn, Bn * WINn);
        // xs = t2 * Wsp + b + x -> bufB
        GEMMF(bufA, Wt_sp, spB, x, bufB, TOK, Dn, Dn, 0);

        // xi permute -> bufA
        permute_xi<<<(int)(XEL / 256), 256, 0, stream>>>(bufB, bufA, ps, pn);
        // intra K/V projections
        GEMMF(bufA, Wt_ik, ikB, nullptr, bufC, TOK, Dn, Dn, 0);
        GEMMF(bufA, Wt_iv, ivB, nullptr, bufD, TOK, Dn, Dn, 0);
        // learned query projection
        GEMMF(quer, Wt_iq, iqB, nullptr, qproj, Nn * pn, Dn, Dn, 0);
        // intra attention -> bufI2 [B*N*pn, D]
        intra_attn<<<Bn * Nn, pn * Hn, 0, stream>>>(qproj, bufC, bufD, bufI2, ps, pn);
        // intra O projection -> bufI2 via bufI1 staging: project then pad-transpose
        GEMMF(bufI2, Wt_io, ioB, nullptr, bufI1, Bn * Nn * pn, Dn, Dn, 0);
        // pad-transpose [.,pn,D] -> [.,D,Kp], then intra_lin (fast path, K=Kp) -> bufE
        tr_io_pad<<<(int)(((size_t)Bn * Nn * Dn * Kp + 255) / 256), 256, 0, stream>>>(
            bufI1, bufI2, pn, Kp);
        GEMMF(bufI2, Wt_ln, lnB, nullptr, bufE, BND, WINn, Kp, 0);

        // inter projections over dm (xe == xi flat [B*N*pn, dm])
        GEMMF(bufA, Wt_q2, q2B, nullptr, bufC, Bn * Nn * pn, dm, dm, 0);
        GEMMF(bufA, Wt_k2, k2B, nullptr, bufD, Bn * Nn * pn, dm, dm, 0);
        GEMMF(bufA, Wt_v2, v2B, nullptr, bufG, Bn * Nn * pn, dm, dm, 0);
        // inter attention -> bufA (xi dead)
        inter_attn<<<dim3(Bn * Nn, Hn), 64, 0, stream>>>(bufC, bufD, bufG, bufA, pn, dm);
        // inter O projection -> bufC
        GEMMF(bufA, Wt_o2, o2B, nullptr, bufC, Bn * Nn * pn, dm, dm, 0);

        // residual combine + LN1 -> bufD
        combine_ln1<<<(TOK + 255) / 256, 256, 0, stream>>>(bufB, bufE, bufC, n1g, n1b,
                                                           bufD, ps, pn);
        // FFN (chunked): gelu(x1*W1+b1)*W2+b2 -> bufG
        for (int c = 0; c < TOK / CHUNK; ++c) {
            const float* xin = bufD + (size_t)c * CHUNK * Dn;
            GEMMF(xin, Wt_f1, f1B, nullptr, bufH, CHUNK, DFFn, Dn, 1);
            GEMMF(bufH, Wt_f2, f2B, nullptr, bufG + (size_t)c * CHUNK * Dn, CHUNK, Dn, DFFn, 0);
        }
        // LN2 + gate-weighted accumulate into d_out
        final_ln2_acc<<<(TOK + 255) / 256, 256, 0, stream>>>(bufD, bufG, n2g, n2b,
                                                             gates, out, e);
    }
}
